// Self_Attn_16174846836787
// MI455X (gfx1250) — matrix operand order repaired and checked by reading, not compiled, not run on hardware
//
#include <hip/hip_runtime.h>
#include <hip/hip_bf16.h>

// Sizes from the reference
#define BB   8
#define CC   512
#define WW   2048
#define CQK  64
#define OUT_ELEMS (BB*CC*WW)   // 8,388,608 floats; attn follows in d_out

typedef _Float16 v16h __attribute__((ext_vector_type(16)));
typedef _Float16 v8h  __attribute__((ext_vector_type(8)));
typedef float    v8f  __attribute__((ext_vector_type(8)));

__device__ __forceinline__ v8f wmma_f16(v16h a, v16h b, v8f c) {
  return __builtin_amdgcn_wmma_f32_16x16x32_f16(false, a, false, b, (short)0, c,
                                                false, false);
}

// Async global->LDS copy, 16 bytes per lane. LDS byte address = low 32 bits of
// the flat pointer. Tracked on ASYNCcnt.
__device__ __forceinline__ void async_b128(void* lds, const void* g) {
  asm volatile("global_load_async_to_lds_b128 %0, %1, off"
               :: "v"((unsigned)(size_t)lds), "v"(g)
               : "memory");
}
__device__ __forceinline__ void wait_async0() {
  asm volatile("s_wait_asynccnt 0" ::: "memory");
}

// A fragment (16x32 f16), tile rows are ldk halves long, K-contiguous.
// lanes 0-15: row m0+lane, halves0-15=K0..15
// lanes16-31: row m0+lane-16, halves0-15=K16..31
__device__ __forceinline__ v16h load_frag_a(const _Float16* As, int ldk, int m0,
                                            int k0, int lane) {
  int row = m0 + (lane & 15);
  int kb  = k0 + ((lane >> 4) << 4);
  v8h lo = *(const v8h*)(As + row * ldk + kb);
  v8h hi = *(const v8h*)(As + row * ldk + kb + 8);
  return __builtin_shufflevector(lo, hi, 0,1,2,3,4,5,6,7,8,9,10,11,12,13,14,15);
}

// B fragment (32x16 f16) from a [N][K] (B-transposed) LDS tile.
// lanes 0-15: col n0+lane, halves = K0..15 ; lanes16-31: K16..31
__device__ __forceinline__ v16h load_frag_b(const _Float16* Bs, int ldk, int n0,
                                            int k0, int lane) {
  int col = n0 + (lane & 15);
  int kb  = k0 + ((lane >> 4) << 4);
  return *(const v16h*)(Bs + col * ldk + kb);
}

__device__ __forceinline__ void cvt16_f32_f16(_Float16* d, const float* s) {
  float4 f0 = ((const float4*)s)[0];
  float4 f1 = ((const float4*)s)[1];
  float4 f2 = ((const float4*)s)[2];
  float4 f3 = ((const float4*)s)[3];
  d[0]=(_Float16)f0.x;  d[1]=(_Float16)f0.y;  d[2]=(_Float16)f0.z;  d[3]=(_Float16)f0.w;
  d[4]=(_Float16)f1.x;  d[5]=(_Float16)f1.y;  d[6]=(_Float16)f1.z;  d[7]=(_Float16)f1.w;
  d[8]=(_Float16)f2.x;  d[9]=(_Float16)f2.y;  d[10]=(_Float16)f2.z; d[11]=(_Float16)f2.w;
  d[12]=(_Float16)f3.x; d[13]=(_Float16)f3.y; d[14]=(_Float16)f3.z; d[15]=(_Float16)f3.w;
}

// ---------------- Kernel 1: x [b][c][w] f32 -> xT [b][w][c] f16 ----------------
__global__ void k_transpose(const float* __restrict__ x, _Float16* __restrict__ xT) {
  __shared__ float tile[32][33];
  int b  = blockIdx.z;
  int w0 = blockIdx.x * 32, c0 = blockIdx.y * 32;
  int tx = threadIdx.x, ty = threadIdx.y;
  const float* xp = x + (size_t)b * CC * WW;
#pragma unroll
  for (int r = 0; r < 4; ++r) {
    int c = c0 + ty + r * 8;
    tile[ty + r * 8][tx] = xp[(size_t)c * WW + w0 + tx];
  }
  __syncthreads();
  _Float16* xo = xT + (size_t)b * WW * CC;
#pragma unroll
  for (int r = 0; r < 4; ++r) {
    int w = w0 + ty + r * 8;
    xo[(size_t)w * CC + c0 + tx] = (_Float16)tile[tx][ty + r * 8];
  }
}

// ------------- Kernel 2: P[b][w][q] = Wp(64x512) @ x^T + bp  (Q and K) ---------
__global__ void k_proj_qk(const _Float16* __restrict__ xT, const float* __restrict__ Wp,
                          const float* __restrict__ bp, _Float16* __restrict__ P) {
  __shared__ _Float16 Ah[128 * 32];  // xT tile [w][c]
  __shared__ _Float16 Bh[64 * 32];   // Wp tile [q][c]
  int b = blockIdx.y, w0 = blockIdx.x * 128;
  int tid = threadIdx.x, lane = tid & 31, wv = tid >> 5;
  int m0 = (wv & 3) * 32, n0 = (wv >> 2) * 32;
  const _Float16* xb = xT + (size_t)b * WW * CC;
  v8f acc[2][2] = {};
  for (int kc = 0; kc < CC; kc += 32) {
    {  // A: 128 rows x 32 halves, async global->LDS (no conversion needed)
      int row = tid >> 1, ch = (tid & 1) * 16;
      const _Float16* s = xb + (size_t)(w0 + row) * CC + kc + ch;
      async_b128(Ah + row * 32 + ch,     s);
      async_b128(Ah + row * 32 + ch + 8, s + 8);
    }
    {  // B: 64 q-rows x 32 f32 -> f16
      int q = tid >> 2, ch = (tid & 3) * 8;
      float4 f0 = ((const float4*)(Wp + (size_t)q * CC + kc + ch))[0];
      float4 f1 = ((const float4*)(Wp + (size_t)q * CC + kc + ch))[1];
      _Float16* d = Bh + q * 32 + ch;
      d[0]=(_Float16)f0.x; d[1]=(_Float16)f0.y; d[2]=(_Float16)f0.z; d[3]=(_Float16)f0.w;
      d[4]=(_Float16)f1.x; d[5]=(_Float16)f1.y; d[6]=(_Float16)f1.z; d[7]=(_Float16)f1.w;
    }
    wait_async0();
    __syncthreads();
    v16h a0 = load_frag_a(Ah, 32, m0,      0, lane);
    v16h a1 = load_frag_a(Ah, 32, m0 + 16, 0, lane);
    v16h b0 = load_frag_b(Bh, 32, n0,      0, lane);
    v16h b1 = load_frag_b(Bh, 32, n0 + 16, 0, lane);
    acc[0][0] = wmma_f16(a0, b0, acc[0][0]);
    acc[0][1] = wmma_f16(a0, b1, acc[0][1]);
    acc[1][0] = wmma_f16(a1, b0, acc[1][0]);
    acc[1][1] = wmma_f16(a1, b1, acc[1][1]);
    __syncthreads();
  }
  _Float16* Pb = P + (size_t)b * WW * CQK;
#pragma unroll
  for (int mt = 0; mt < 2; ++mt)
#pragma unroll
    for (int nt = 0; nt < 2; ++nt) {
      int col = n0 + nt * 16 + (lane & 15);
      float bias = bp[col];
#pragma unroll
      for (int r = 0; r < 8; ++r) {
        int row = w0 + m0 + mt * 16 + r + ((lane >> 4) << 3);
        Pb[(size_t)row * CQK + col] = (_Float16)(acc[mt][nt][r] + bias);
      }
    }
}

// ------------- Kernel 3: V[b][c][w] = Wv(512x512) @ x + bv  (f16 out) ----------
__global__ void k_proj_v(const _Float16* __restrict__ xT, const float* __restrict__ Wv,
                         const float* __restrict__ bv, _Float16* __restrict__ V) {
  __shared__ _Float16 Ah[128 * 32];  // Wv tile [c][k]
  __shared__ _Float16 Bh[128 * 32];  // xT tile [w][k]
  int b = blockIdx.z, c0 = blockIdx.x * 128, w0 = blockIdx.y * 128;
  int tid = threadIdx.x, lane = tid & 31, wv = tid >> 5;
  int m0 = (wv & 3) * 32, n0 = (wv >> 2) * 64;
  const _Float16* xb = xT + (size_t)b * WW * CC;
  v8f acc[2][4] = {};
  for (int kc = 0; kc < CC; kc += 32) {
    {  // A: Wv rows f32 -> f16 (must pass through VGPRs)
      int row = tid >> 1, ch = (tid & 1) * 16;
      cvt16_f32_f16(Ah + row * 32 + ch, Wv + (size_t)(c0 + row) * CC + kc + ch);
    }
    {  // B: xT rows f16, async global->LDS
      int row = tid >> 1, ch = (tid & 1) * 16;
      const _Float16* s = xb + (size_t)(w0 + row) * CC + kc + ch;
      async_b128(Bh + row * 32 + ch,     s);
      async_b128(Bh + row * 32 + ch + 8, s + 8);
    }
    wait_async0();
    __syncthreads();
    v16h a0 = load_frag_a(Ah, 32, m0,      0, lane);
    v16h a1 = load_frag_a(Ah, 32, m0 + 16, 0, lane);
#pragma unroll
    for (int nt = 0; nt < 4; ++nt) {
      v16h bf = load_frag_b(Bh, 32, n0 + nt * 16, 0, lane);
      acc[0][nt] = wmma_f16(a0, bf, acc[0][nt]);
      acc[1][nt] = wmma_f16(a1, bf, acc[1][nt]);
    }
    __syncthreads();
  }
  _Float16* Vb = V + (size_t)b * CC * WW;
#pragma unroll
  for (int mt = 0; mt < 2; ++mt)
#pragma unroll
    for (int nt = 0; nt < 4; ++nt) {
      int col = w0 + n0 + nt * 16 + (lane & 15);
#pragma unroll
      for (int r = 0; r < 8; ++r) {
        int c = c0 + m0 + mt * 16 + r + ((lane >> 4) << 3);
        Vb[(size_t)c * WW + col] = (_Float16)(acc[mt][nt][r] + bv[c]);
      }
    }
}

// ------------- Kernel 4: energy[b][i][j] = Q[i][:] . K[j][:]  (K-dim 64) -------
__global__ void k_energy(const _Float16* __restrict__ Qh, const _Float16* __restrict__ Kh,
                         float* __restrict__ attn) {
  __shared__ _Float16 Ah[128 * 64];
  __shared__ _Float16 Bh[128 * 64];
  int b = blockIdx.z, i0 = blockIdx.x * 128, j0 = blockIdx.y * 128;
  int tid = threadIdx.x, lane = tid & 31, wv = tid >> 5;
  int m0 = (wv & 3) * 32, n0 = (wv >> 2) * 64;
  const _Float16* Qb = Qh + (size_t)b * WW * CQK;
  const _Float16* Kb = Kh + (size_t)b * WW * CQK;
  {  // both tiles are raw f16 copies -> async global->LDS
    int row = tid >> 1, ch = (tid & 1) * 32;
    const _Float16* sa = Qb + (size_t)(i0 + row) * CQK + ch;
    const _Float16* sb = Kb + (size_t)(j0 + row) * CQK + ch;
#pragma unroll
    for (int t = 0; t < 4; ++t) {
      async_b128(Ah + row * 64 + ch + t * 8, sa + t * 8);
      async_b128(Bh + row * 64 + ch + t * 8, sb + t * 8);
    }
  }
  wait_async0();
  __syncthreads();
  v8f acc[2][4] = {};
#pragma unroll
  for (int k0 = 0; k0 < 64; k0 += 32) {
    v16h a0 = load_frag_a(Ah, 64, m0,      k0, lane);
    v16h a1 = load_frag_a(Ah, 64, m0 + 16, k0, lane);
#pragma unroll
    for (int nt = 0; nt < 4; ++nt) {
      v16h bf = load_frag_b(Bh, 64, n0 + nt * 16, k0, lane);
      acc[0][nt] = wmma_f16(a0, bf, acc[0][nt]);
      acc[1][nt] = wmma_f16(a1, bf, acc[1][nt]);
    }
  }
  float* ab = attn + (size_t)b * WW * WW;
#pragma unroll
  for (int mt = 0; mt < 2; ++mt)
#pragma unroll
    for (int nt = 0; nt < 4; ++nt) {
      int col = j0 + n0 + nt * 16 + (lane & 15);
#pragma unroll
      for (int r = 0; r < 8; ++r) {
        int row = i0 + m0 + mt * 16 + r + ((lane >> 4) << 3);
        ab[(size_t)row * WW + col] = acc[mt][nt][r];
      }
    }
}

// ------------- Kernel 5: row-wise softmax in place (one wave per row) ----------
__global__ void k_softmax(float* __restrict__ attn) {
  int row  = blockIdx.x * 8 + (threadIdx.x >> 5);
  int lane = threadIdx.x & 31;
  float4* p = (float4*)(attn + (size_t)row * WW);
  float4 v[16];
  float m = -1e30f;
#pragma unroll
  for (int t = 0; t < 16; ++t) {
    v[t] = p[t * 32 + lane];
    m = fmaxf(m, fmaxf(fmaxf(v[t].x, v[t].y), fmaxf(v[t].z, v[t].w)));
  }
#pragma unroll
  for (int off = 16; off > 0; off >>= 1) m = fmaxf(m, __shfl_xor(m, off, 32));
  float s = 0.f;
#pragma unroll
  for (int t = 0; t < 16; ++t) {
    v[t].x = __expf(v[t].x - m); v[t].y = __expf(v[t].y - m);
    v[t].z = __expf(v[t].z - m); v[t].w = __expf(v[t].w - m);
    s += v[t].x + v[t].y + v[t].z + v[t].w;
  }
#pragma unroll
  for (int off = 16; off > 0; off >>= 1) s += __shfl_xor(s, off, 32);
  float inv = 1.f / s;
#pragma unroll
  for (int t = 0; t < 16; ++t) {
    v[t].x *= inv; v[t].y *= inv; v[t].z *= inv; v[t].w *= inv;
    p[t * 32 + lane] = v[t];
  }
}

// ------- Kernel 6: out[b][c][i] = gamma * sum_j V[c][j]*attn[i][j] + x ---------
// K-chunk 64: 16 WMMAs per wave per barrier pair.
__global__ void k_out(const _Float16* __restrict__ Vh, const float* __restrict__ attn,
                      const float* __restrict__ x, const float* __restrict__ gamma,
                      float* __restrict__ out) {
  __shared__ _Float16 Ah[128 * 64];  // V tile [c][j]   (async f16 copy)
  __shared__ _Float16 Bh[128 * 64];  // attn tile [i][j] (f32 -> f16)
  int b = blockIdx.z, c0 = blockIdx.x * 128, i0 = blockIdx.y * 128;
  int tid = threadIdx.x, lane = tid & 31, wv = tid >> 5;
  int m0 = (wv & 3) * 32, n0 = (wv >> 2) * 64;
  const _Float16* Vb = Vh + (size_t)b * CC * WW;
  const float*    ab = attn + (size_t)b * WW * WW;
  v8f acc[2][4] = {};
  for (int jc = 0; jc < WW; jc += 64) {
    int row = tid >> 1, ch = (tid & 1) * 32;
    {  // A: V f16, async global->LDS (4 x b128 per thread)
      const _Float16* s = Vb + (size_t)(c0 + row) * WW + jc + ch;
#pragma unroll
      for (int t = 0; t < 4; ++t) async_b128(Ah + row * 64 + ch + t * 8, s + t * 8);
    }
    {  // B: attn f32 -> f16 via VGPRs
      const float* s = ab + (size_t)(i0 + row) * WW + jc + ch;
      cvt16_f32_f16(Bh + row * 64 + ch,      s);
      cvt16_f32_f16(Bh + row * 64 + ch + 16, s + 16);
    }
    wait_async0();
    __syncthreads();
#pragma unroll
    for (int k0 = 0; k0 < 64; k0 += 32) {
      v16h a0 = load_frag_a(Ah, 64, m0,      k0, lane);
      v16h a1 = load_frag_a(Ah, 64, m0 + 16, k0, lane);
#pragma unroll
      for (int nt = 0; nt < 4; ++nt) {
        v16h bf = load_frag_b(Bh, 64, n0 + nt * 16, k0, lane);
        acc[0][nt] = wmma_f16(a0, bf, acc[0][nt]);
        acc[1][nt] = wmma_f16(a1, bf, acc[1][nt]);
      }
    }
    __syncthreads();
  }
  float g = gamma[0];
#pragma unroll
  for (int mt = 0; mt < 2; ++mt)
#pragma unroll
    for (int nt = 0; nt < 4; ++nt) {
      int col = i0 + n0 + nt * 16 + (lane & 15);
#pragma unroll
      for (int r = 0; r < 8; ++r) {
        int c = c0 + m0 + mt * 16 + r + ((lane >> 4) << 3);
        size_t idx = (size_t)b * CC * WW + (size_t)c * WW + col;
        out[idx] = g * acc[mt][nt][r] + x[idx];
      }
    }
}

extern "C" void kernel_launch(void* const* d_in, const int* in_sizes, int n_in,
                              void* d_out, int out_size, void* d_ws, size_t ws_size,
                              hipStream_t stream) {
  const float* x     = (const float*)d_in[0];
  const float* Wq    = (const float*)d_in[1];
  const float* bq    = (const float*)d_in[2];
  const float* Wk    = (const float*)d_in[3];
  const float* bk    = (const float*)d_in[4];
  const float* Wv    = (const float*)d_in[5];
  const float* bv    = (const float*)d_in[6];
  const float* gamma = (const float*)d_in[7];

  float* out  = (float*)d_out;
  float* attn = out + OUT_ELEMS;  // (B, W, W) region of d_out

  // Workspace: xT (16MB) | Q (2MB) | K (2MB) | V (16MB) = 36MB f16
  _Float16* xT = (_Float16*)d_ws;
  _Float16* Qh = xT + (size_t)BB * WW * CC;
  _Float16* Kh = Qh + (size_t)BB * WW * CQK;
  _Float16* Vh = Kh + (size_t)BB * WW * CQK;

  k_transpose<<<dim3(WW / 32, CC / 32, BB), dim3(32, 8), 0, stream>>>(x, xT);
  k_proj_qk<<<dim3(WW / 128, BB), 256, 0, stream>>>(xT, Wq, bq, Qh);
  k_proj_qk<<<dim3(WW / 128, BB), 256, 0, stream>>>(xT, Wk, bk, Kh);
  k_proj_v<<<dim3(CC / 128, WW / 128, BB), 256, 0, stream>>>(xT, Wv, bv, Vh);
  k_energy<<<dim3(WW / 128, WW / 128, BB), 256, 0, stream>>>(Qh, Kh, attn);
  k_softmax<<<(BB * WW) / 8, 256, 0, stream>>>(attn);
  k_out<<<dim3(CC / 128, WW / 128, BB), 256, 0, stream>>>(Vh, attn, x, gamma, out);
}
